// FastTransformerBlock_57440892617544
// MI455X (gfx1250) — matrix operand order repaired and checked by reading, not compiled, not run on hardware
//
#include <hip/hip_runtime.h>
#include <hip/hip_bf16.h>

// ---------------------------------------------------------------------------
// FastTransformerBlock (6-layer linear-attention encoder) for MI455X gfx1250.
// GEMMs run on v_wmma_f32_16x16x32_bf16 (f32 accumulate), staging f32->bf16
// into LDS. Attention einsums + LayerNorm are small VALU kernels.
// ---------------------------------------------------------------------------

#define LNUM   6
#define HEADS  16
#define DMODEL 1024
#define DFFN   4096
#define BATCH  4
#define SEQ    2048
#define DH     64
#define EPS_ATTN 1e-6f
#define EPS_LN   1e-5f

typedef __attribute__((ext_vector_type(16))) __bf16 v16bf;
typedef __attribute__((ext_vector_type(8)))  float  v8f;

__device__ __forceinline__ __bf16 to_bf16(float f) { return (__bf16)f; }

// ---------------------------------------------------------------------------
// GEMM: C[M,N] = act(A[M,K] @ B[K,N] + bias[N] (+ res[M,N]))
// act: 0 = none, 1 = elu(x)+1 (== x>0 ? x+1 : exp(x)), 2 = relu
// Block tile 128x128, K-step 32. 8 wave32s; each wave computes 32x64
// via 2x4 WMMA fragments. All of M,N divisible by 128; K divisible by 32.
// ---------------------------------------------------------------------------
#define BM 128
#define BN 128
#define BK 32
#define LDSP 40   // padded per-row stride (halves)

__global__ __launch_bounds__(256)
void gemm_bf16_wmma(const float* __restrict__ A, const float* __restrict__ B,
                    const float* __restrict__ bias, const float* __restrict__ res,
                    float* __restrict__ C, int M, int N, int K, int act)
{
    __shared__ __bf16 As[BM][LDSP];   // [m][k]
    __shared__ __bf16 Bs[BN][LDSP];   // [n][k] (transposed on store)

    const int tid  = threadIdx.x;
    const int lane = tid & 31;
    const int wave = tid >> 5;
    const int wr   = wave & 3;     // wave row in M (0..3) -> 32 rows each
    const int wc   = wave >> 2;    // wave col in N (0..1) -> 64 cols each
    const int m0   = blockIdx.y * BM;
    const int n0   = blockIdx.x * BN;

    const int l16      = lane & 15;
    const bool loHalf  = (lane < 16);
    const int kaBase   = loHalf ? 0 : 16;   // A frag: K 0-15 vs 16-31
    const int kbBase   = loHalf ? 0 : 16;   // B frag: K 0-15 vs 16-31

    v8f acc[2][4] = {};

    for (int k0 = 0; k0 < K; k0 += BK) {
        const int kpf = (k0 + BK < K) ? (k0 + BK) : k0;   // next-tile prefetch
        // ---- stage A tile (128x32 f32 -> bf16), 4 float4 per thread ----
        #pragma unroll
        for (int i = 0; i < 4; ++i) {
            int f   = tid + i * 256;          // 0..1023
            int row = f >> 3;                 // 0..127
            int col = (f & 7) << 2;           // 0..28
            const float4 v = *(const float4*)(A + (size_t)(m0 + row) * K + k0 + col);
            __builtin_prefetch(A + (size_t)(m0 + row) * K + kpf + col, 0, 1);
            As[row][col + 0] = to_bf16(v.x);
            As[row][col + 1] = to_bf16(v.y);
            As[row][col + 2] = to_bf16(v.z);
            As[row][col + 3] = to_bf16(v.w);
        }
        // ---- stage B tile (32x128 f32 -> bf16, transposed to [n][k]) ----
        #pragma unroll
        for (int i = 0; i < 4; ++i) {
            int f    = tid + i * 256;         // 0..1023
            int krow = f >> 5;                // 0..31
            int col  = (f & 31) << 2;         // 0..124
            const float4 v = *(const float4*)(B + (size_t)(k0 + krow) * N + n0 + col);
            __builtin_prefetch(B + (size_t)(kpf + krow) * N + n0 + col, 0, 1);
            Bs[col + 0][krow] = to_bf16(v.x);
            Bs[col + 1][krow] = to_bf16(v.y);
            Bs[col + 2][krow] = to_bf16(v.z);
            Bs[col + 3][krow] = to_bf16(v.w);
        }
        __syncthreads();

        // ---- load fragments ----
        v16bf afrag[2];
        #pragma unroll
        for (int fm = 0; fm < 2; ++fm) {
            int row = wr * 32 + fm * 16 + l16;
            #pragma unroll
            for (int i = 0; i < 8; ++i) {
                afrag[fm][i]     = As[row][kaBase + i];
                afrag[fm][8 + i] = As[row][kaBase + 8 + i];
            }
        }
        v16bf bfrag[4];
        #pragma unroll
        for (int fn = 0; fn < 4; ++fn) {
            int col = wc * 64 + fn * 16 + l16;
            #pragma unroll
            for (int i = 0; i < 16; ++i)
                bfrag[fn][i] = Bs[col][kbBase + i];
        }

        // ---- 2x4 WMMA ----
        #pragma unroll
        for (int fm = 0; fm < 2; ++fm)
            #pragma unroll
            for (int fn = 0; fn < 4; ++fn)
                acc[fm][fn] = __builtin_amdgcn_wmma_f32_16x16x32_bf16(
                    false, afrag[fm], false, bfrag[fn],
                    (short)0, acc[fm][fn], false, false);
        __syncthreads();
    }

    // ---- epilogue: bias (+residual) + activation ----
    const bool hasRes = (res != nullptr);
    const int rowOff  = loHalf ? 0 : 8;
    #pragma unroll
    for (int fm = 0; fm < 2; ++fm) {
        #pragma unroll
        for (int fn = 0; fn < 4; ++fn) {
            const int nIdx  = n0 + wc * 64 + fn * 16 + l16;
            const int mBase = m0 + wr * 32 + fm * 16 + rowOff;
            const float bb  = bias[nIdx];
            #pragma unroll
            for (int r = 0; r < 8; ++r) {
                size_t off = (size_t)(mBase + r) * N + nIdx;
                float v = acc[fm][fn][r] + bb;
                if (hasRes) v += res[off];
                if (act == 1)      v = (v > 0.f) ? (v + 1.f) : __expf(v);
                else if (act == 2) v = (v > 0.f) ? v : 0.f;
                C[off] = v;
            }
        }
    }
}

// ---------------------------------------------------------------------------
// kv[b,h] = sum_s k[b,s,h,:] (x) v[b,s,h,:]   (64x64 per head)
// ksum[b,h,:] = sum_s k[b,s,h,:]
// One block (256 thr) per (b,h); each thread owns a 4x4 patch of kv.
// ---------------------------------------------------------------------------
__global__ __launch_bounds__(256)
void linattn_kv(const float* __restrict__ Kf, const float* __restrict__ Vf,
                float* __restrict__ kv, float* __restrict__ ksum)
{
    __shared__ float Ks[32][DH];
    __shared__ float Vs[32][DH];
    const int b = blockIdx.x / HEADS;
    const int h = blockIdx.x % HEADS;
    const int tid = threadIdx.x;
    const int td = tid >> 4;       // 0..15 -> rows d = 4*td..
    const int tm = tid & 15;       // 0..15 -> cols m = 4*tm..

    float acc[4][4] = {};

    for (int s0 = 0; s0 < SEQ; s0 += 32) {
        #pragma unroll
        for (int i = 0; i < 2; ++i) {
            int f   = tid + i * 256;          // 0..511
            int row = f >> 4;                 // 0..31
            int col = (f & 15) << 2;          // 0..60
            size_t g = ((size_t)(b * SEQ + s0 + row)) * DMODEL + h * DH + col;
            *(float4*)&Ks[row][col] = *(const float4*)(Kf + g);
            *(float4*)&Vs[row][col] = *(const float4*)(Vf + g);
        }
        __syncthreads();
        #pragma unroll 4
        for (int j = 0; j < 32; ++j) {
            float kd[4], vm[4];
            #pragma unroll
            for (int r = 0; r < 4; ++r) kd[r] = Ks[j][td * 4 + r];
            #pragma unroll
            for (int c = 0; c < 4; ++c) vm[c] = Vs[j][tm * 4 + c];
            #pragma unroll
            for (int r = 0; r < 4; ++r)
                #pragma unroll
                for (int c = 0; c < 4; ++c)
                    acc[r][c] += kd[r] * vm[c];
        }
        __syncthreads();
    }

    size_t base = (size_t)blockIdx.x * DH * DH;
    #pragma unroll
    for (int r = 0; r < 4; ++r)
        #pragma unroll
        for (int c = 0; c < 4; ++c)
            kv[base + (size_t)(td * 4 + r) * DH + tm * 4 + c] = acc[r][c];

    if (tid < DH) {
        float s = 0.f;
        for (int s1 = 0; s1 < SEQ; ++s1)
            s += Kf[((size_t)(b * SEQ + s1)) * DMODEL + h * DH + tid];
        ksum[(size_t)blockIdx.x * DH + tid] = s;
    }
}

// ---------------------------------------------------------------------------
// o[b,s,h,m] = (q[b,s,h,:] . kv[b,h,:,m]) / (q[b,s,h,:] . ksum[b,h,:] + eps)
// One 64-thread block per (b,s,h).
// ---------------------------------------------------------------------------
__global__ __launch_bounds__(64)
void linattn_o(const float* __restrict__ Qf, const float* __restrict__ kv,
               const float* __restrict__ ksum, float* __restrict__ O)
{
    __shared__ float qs[DH];
    const int idx = blockIdx.x;
    const int h = idx % HEADS;
    const int s = (idx / HEADS) % SEQ;
    const int b = idx / (HEADS * SEQ);
    const int m = threadIdx.x;

    const size_t qoff = ((size_t)(b * SEQ + s)) * DMODEL + h * DH;
    qs[m] = Qf[qoff + m];
    __syncthreads();

    const float* kvp = kv   + ((size_t)(b * HEADS + h)) * DH * DH;
    const float* ksp = ksum + ((size_t)(b * HEADS + h)) * DH;
    float den = EPS_ATTN, o = 0.f;
    #pragma unroll 8
    for (int d = 0; d < DH; ++d) {
        float q = qs[d];
        den += q * ksp[d];
        o   += q * kvp[(size_t)d * DH + m];
    }
    O[qoff + m] = o / den;
}

// ---------------------------------------------------------------------------
// LayerNorm over last dim (D=1024). One 256-thread block per row (4 elem/thr).
// ---------------------------------------------------------------------------
__global__ __launch_bounds__(256)
void layernorm_k(const float* __restrict__ X, const float* __restrict__ w,
                 const float* __restrict__ bb, float* __restrict__ Y)
{
    __shared__ float red[256];
    const int row = blockIdx.x;
    const int t = threadIdx.x;
    const float* xr = X + (size_t)row * DMODEL;

    float4 v = *(const float4*)(xr + t * 4);
    red[t] = v.x + v.y + v.z + v.w;
    __syncthreads();
    #pragma unroll
    for (int s = 128; s > 0; s >>= 1) {
        if (t < s) red[t] += red[t + s];
        __syncthreads();
    }
    const float mean = red[0] * (1.f / DMODEL);
    __syncthreads();

    float dx = v.x - mean, dy = v.y - mean, dz = v.z - mean, dw = v.w - mean;
    red[t] = dx * dx + dy * dy + dz * dz + dw * dw;
    __syncthreads();
    #pragma unroll
    for (int s = 128; s > 0; s >>= 1) {
        if (t < s) red[t] += red[t + s];
        __syncthreads();
    }
    const float rstd = rsqrtf(red[0] * (1.f / DMODEL) + EPS_LN);

    float4 wv = *(const float4*)(w + t * 4);
    float4 bv = *(const float4*)(bb + t * 4);
    float4 o;
    o.x = dx * rstd * wv.x + bv.x;
    o.y = dy * rstd * wv.y + bv.y;
    o.z = dz * rstd * wv.z + bv.z;
    o.w = dw * rstd * wv.w + bv.w;
    *(float4*)(Y + (size_t)row * DMODEL + t * 4) = o;
}

// ---------------------------------------------------------------------------
// Launcher
// ---------------------------------------------------------------------------
extern "C" void kernel_launch(void* const* d_in, const int* in_sizes, int n_in,
                              void* d_out, int out_size, void* d_ws, size_t ws_size,
                              hipStream_t stream) {
    (void)in_sizes; (void)n_in; (void)out_size; (void)ws_size;

    const float* x_in = (const float*)d_in[0];
    const float* Wq   = (const float*)d_in[1];
    const float* bq   = (const float*)d_in[2];
    const float* Wk   = (const float*)d_in[3];
    const float* bk   = (const float*)d_in[4];
    const float* Wv   = (const float*)d_in[5];
    const float* bv   = (const float*)d_in[6];
    const float* Wo   = (const float*)d_in[7];
    const float* bo   = (const float*)d_in[8];
    const float* ln1w = (const float*)d_in[9];
    const float* ln1b = (const float*)d_in[10];
    const float* W1   = (const float*)d_in[11];
    const float* b1   = (const float*)d_in[12];
    const float* W2   = (const float*)d_in[13];
    const float* b2   = (const float*)d_in[14];
    const float* ln2w = (const float*)d_in[15];
    const float* ln2b = (const float*)d_in[16];
    const float* lnfw = (const float*)d_in[17];
    const float* lnfb = (const float*)d_in[18];

    // Workspace layout (all f32):
    //   X   [B*S*D]              running activations
    //   T   [B*S*D]              pre-norm temp (residual sums)
    //   BIG [B*S*DFF]            holds {Q,K,V,O} during attention, H during FFN
    //   kv  [B*H*64*64], ksum [B*H*64]
    char* ws = (char*)d_ws;
    const size_t SZ = (size_t)BATCH * SEQ * DMODEL * sizeof(float);  // 33.5 MB
    float* X   = (float*)(ws);
    float* T   = (float*)(ws + SZ);
    char*  big = ws + 2 * SZ;                       // 4*SZ bytes
    float* Q   = (float*)(big);
    float* Kb  = (float*)(big + SZ);
    float* Vb  = (float*)(big + 2 * SZ);
    float* Ob  = (float*)(big + 3 * SZ);
    float* Hb  = (float*)(big);                     // overlaps Q..O (dead then)
    float* kvb = (float*)(ws + 6 * SZ);
    float* ksb = kvb + (size_t)BATCH * HEADS * DH * DH;

    hipMemcpyAsync(X, x_in, SZ, hipMemcpyDeviceToDevice, stream);

    const int Mrows = BATCH * SEQ;                  // 8192
    dim3 blk(256);
    dim3 gD(DMODEL / BN, Mrows / BM);               // 8 x 64
    dim3 gF(DFFN / BN,   Mrows / BM);               // 32 x 64
    dim3 gLN(Mrows);

    for (int i = 0; i < LNUM; ++i) {
        const float* wq = Wq + (size_t)i * DMODEL * DMODEL;
        const float* wk = Wk + (size_t)i * DMODEL * DMODEL;
        const float* wv = Wv + (size_t)i * DMODEL * DMODEL;
        const float* wo = Wo + (size_t)i * DMODEL * DMODEL;
        const float* w1 = W1 + (size_t)i * DMODEL * DFFN;
        const float* w2 = W2 + (size_t)i * DFFN * DMODEL;

        // Q = phi(X@Wq+bq), K = phi(X@Wk+bk), V = X@Wv+bv
        gemm_bf16_wmma<<<gD, blk, 0, stream>>>(X, wq, bq + i * DMODEL, nullptr,
                                               Q,  Mrows, DMODEL, DMODEL, 1);
        gemm_bf16_wmma<<<gD, blk, 0, stream>>>(X, wk, bk + i * DMODEL, nullptr,
                                               Kb, Mrows, DMODEL, DMODEL, 1);
        gemm_bf16_wmma<<<gD, blk, 0, stream>>>(X, wv, bv + i * DMODEL, nullptr,
                                               Vb, Mrows, DMODEL, DMODEL, 0);
        // linear attention
        linattn_kv<<<BATCH * HEADS, 256, 0, stream>>>(Kb, Vb, kvb, ksb);
        linattn_o<<<BATCH * SEQ * HEADS, 64, 0, stream>>>(Q, kvb, ksb, Ob);
        // T = O@Wo + bo + X  (residual fused)
        gemm_bf16_wmma<<<gD, blk, 0, stream>>>(Ob, wo, bo + i * DMODEL, X,
                                               T,  Mrows, DMODEL, DMODEL, 0);
        // X = LN1(T)
        layernorm_k<<<gLN, blk, 0, stream>>>(T, ln1w + i * DMODEL,
                                             ln1b + i * DMODEL, X);
        // H = relu(X@W1 + b1)
        gemm_bf16_wmma<<<gF, blk, 0, stream>>>(X, w1, b1 + i * DFFN, nullptr,
                                               Hb, Mrows, DFFN, DMODEL, 2);
        // T = H@W2 + b2 + X
        gemm_bf16_wmma<<<gD, blk, 0, stream>>>(Hb, w2, b2 + i * DMODEL, X,
                                               T,  Mrows, DMODEL, DFFN, 0);
        // X = LN2(T)
        layernorm_k<<<gLN, blk, 0, stream>>>(T, ln2w + i * DMODEL,
                                             ln2b + i * DMODEL, X);
    }
    // final LayerNorm -> d_out
    layernorm_k<<<gLN, blk, 0, stream>>>(X, lnfw, lnfb, (float*)d_out);
}
